// BWCholeskyBlock_47064251630241
// MI455X (gfx1250) — matrix operand order repaired and checked by reading, not compiled, not run on hardware
//
#include <hip/hip_runtime.h>
#include <hip/hip_bf16.h>

// ---------------------------------------------------------------------------
// cholesky_batch whitening for MI455X (gfx1250, wave32, WMMA)
//
//   mean -> xc -> cov = xc^T xc / N (+eps I, lower triangle only)
//   L = chol(cov) -> W = L^{-1} -> Y = xc * W^T + beta
//
// GEMMs run on v_wmma_f32_16x16x32_bf16 with the 3-term bf16 split
// (hi*hi + hi*lo + lo*hi, f32 accumulate) for ~fp32 accuracy at bf16
// matrix-pipe rates. LDS tiles are K-contiguous per row: every fragment is
// two ds_load_b128; staging stores are packed ds_store_b64 (register 4x4
// transpose in the SYRK). SYRK skips blocks above the diagonal (Cholesky
// reads lower only). X (128 MB) is L2-resident (192 MB L2).
// ---------------------------------------------------------------------------

typedef __attribute__((ext_vector_type(16))) __bf16        v16bf;
typedef __attribute__((ext_vector_type(8)))  float         v8f;
typedef __attribute__((ext_vector_type(4)))  float         f32x4;
typedef __attribute__((ext_vector_type(2)))  unsigned int  u32x2;
typedef __attribute__((ext_vector_type(4)))  unsigned int  u32x4;
typedef __attribute__((ext_vector_type(8)))  unsigned int  u32x8;

#define BN_EPS  1e-5f
#define LDS_LD  40   // row stride in bf16: 80B, 16B-aligned, conflict-free b128

__device__ __forceinline__ void bf16split(float v, __bf16& hi, __bf16& lo) {
    hi = (__bf16)v;
    lo = (__bf16)(v - (float)hi);
}

__device__ __forceinline__ unsigned int bpack2(__bf16 a, __bf16 b) {
    union { __bf16 h[2]; unsigned int u; } x;
    x.h[0] = a; x.h[1] = b;
    return x.u;
}
// one ds_store_b64: 4 contiguous bf16
__device__ __forceinline__ void store4bf(__bf16* dst, __bf16 a, __bf16 b,
                                         __bf16 c, __bf16 d) {
    u32x2 v; v[0] = bpack2(a, b); v[1] = bpack2(c, d);
    *(u32x2*)dst = v;
}

// A operand (16x32 bf16): lane reads K = koff..koff+7 and koff+16..koff+23,
// koff = 8*(lane>=16). Two 16B contiguous LDS loads.
__device__ __forceinline__ v16bf load_fragA(const __bf16* row, int lane) {
    const int koff = (lane >> 4) << 3;
    u32x4 c0 = *(const u32x4*)(row + koff);
    u32x4 c1 = *(const u32x4*)(row + koff + 16);
    u32x8 u;
    u[0] = c0[0]; u[1] = c0[1]; u[2] = c0[2]; u[3] = c0[3];
    u[4] = c1[0]; u[5] = c1[1]; u[6] = c1[2]; u[7] = c1[3];
    return __builtin_bit_cast(v16bf, u);
}

// B operand (32x16 bf16): lane reads K = koff..koff+7 and koff+16..koff+23,
// koff = 8*(lane>=16). Two 16B contiguous LDS loads.
__device__ __forceinline__ v16bf load_fragB(const __bf16* row, int lane) {
    const int koff = (lane >> 4) << 3;
    u32x4 c0 = *(const u32x4*)(row + koff);
    u32x4 c1 = *(const u32x4*)(row + koff + 16);
    u32x8 u;
    u[0] = c0[0]; u[1] = c0[1]; u[2] = c0[2]; u[3] = c0[3];
    u[4] = c1[0]; u[5] = c1[1]; u[6] = c1[2]; u[7] = c1[3];
    return __builtin_bit_cast(v16bf, u);
}

// ---------------------------------------------------------------- utilities
__global__ __launch_bounds__(256) void zero_f32(float* p, long n) {
    long i = (long)blockIdx.x * blockDim.x + threadIdx.x;
    long stride = (long)gridDim.x * blockDim.x;
    for (; i < n; i += stride) p[i] = 0.0f;
}

// ------------------------------------------------------------- column mean
__global__ __launch_bounds__(256) void mean_partial(const float* __restrict__ X,
                                                    float* __restrict__ part,
                                                    int F, int rowsPer) {
    int f  = blockIdx.x * 256 + threadIdx.x;
    int r0 = blockIdx.y * rowsPer;
    float s = 0.0f;
    for (int r = 0; r < rowsPer; ++r)
        s += X[(size_t)(r0 + r) * F + f];
    part[(size_t)blockIdx.y * F + f] = s;
}

__global__ __launch_bounds__(256) void mean_reduce(const float* __restrict__ part,
                                                   float* __restrict__ mu,
                                                   int F, int G, float invN) {
    int f = blockIdx.x * 256 + threadIdx.x;
    if (f >= F) return;
    float s = 0.0f;
    for (int g = 0; g < G; ++g) s += part[(size_t)g * F + f];
    mu[f] = s * invN;
}

// ------------------------------------------------- cov = xc^T xc / N + eps I
// 128(i) x 64(j) tile per block; blocks strictly above the diagonal are
// skipped (only the lower triangle is consumed downstream). Staging does a
// 4x4 register transpose so all LDS writes are packed ds_store_b64.
__global__ __launch_bounds__(256) void cov_syrk_wmma(const float* __restrict__ X,
                                                     const float* __restrict__ mu,
                                                     float* __restrict__ C,
                                                     int N, int F) {
    __shared__ __attribute__((aligned(16))) __bf16 Ah[128][LDS_LD], Al[128][LDS_LD];
    __shared__ __attribute__((aligned(16))) __bf16 Bh[64][LDS_LD],  Bl[64][LDS_LD];
    const int i0 = blockIdx.y * 128, j0 = blockIdx.x * 64;
    if (j0 >= i0 + 128) return;                 // above diagonal: not needed
    const int tid = threadIdx.x, lane = tid & 31, wid = tid >> 5;

    v8f acc[4] = {};

    for (int k0 = 0; k0 < N; k0 += 32) {
        __syncthreads();
        // stage A: X[k0+r][i0+c] -> A[c][r]; 4 samples x 4 features per thread
        {
            const int r0 = (tid >> 5) << 2;     // sample quad 0..28
            const int c0 = (tid & 31) << 2;     // feature quad 0..124
            f32x4 x0 = *(const f32x4*)&X[(size_t)(k0 + r0 + 0) * F + i0 + c0];
            f32x4 x1 = *(const f32x4*)&X[(size_t)(k0 + r0 + 1) * F + i0 + c0];
            f32x4 x2 = *(const f32x4*)&X[(size_t)(k0 + r0 + 2) * F + i0 + c0];
            f32x4 x3 = *(const f32x4*)&X[(size_t)(k0 + r0 + 3) * F + i0 + c0];
            f32x4 m  = *(const f32x4*)&mu[i0 + c0];
            #pragma unroll
            for (int t = 0; t < 4; ++t) {
                __bf16 h0, h1, h2, h3, l0, l1, l2, l3;
                bf16split(x0[t] - m[t], h0, l0);
                bf16split(x1[t] - m[t], h1, l1);
                bf16split(x2[t] - m[t], h2, l2);
                bf16split(x3[t] - m[t], h3, l3);
                store4bf(&Ah[c0 + t][r0], h0, h1, h2, h3);
                store4bf(&Al[c0 + t][r0], l0, l1, l2, l3);
            }
        }
        // stage B: X[k0+r][j0+c] -> B[c][r]; 128 groups (half the threads)
        if (tid < 128) {
            const int r0 = (tid >> 4) << 2;
            const int c0 = (tid & 15) << 2;
            f32x4 x0 = *(const f32x4*)&X[(size_t)(k0 + r0 + 0) * F + j0 + c0];
            f32x4 x1 = *(const f32x4*)&X[(size_t)(k0 + r0 + 1) * F + j0 + c0];
            f32x4 x2 = *(const f32x4*)&X[(size_t)(k0 + r0 + 2) * F + j0 + c0];
            f32x4 x3 = *(const f32x4*)&X[(size_t)(k0 + r0 + 3) * F + j0 + c0];
            f32x4 m  = *(const f32x4*)&mu[j0 + c0];
            #pragma unroll
            for (int t = 0; t < 4; ++t) {
                __bf16 h0, h1, h2, h3, l0, l1, l2, l3;
                bf16split(x0[t] - m[t], h0, l0);
                bf16split(x1[t] - m[t], h1, l1);
                bf16split(x2[t] - m[t], h2, l2);
                bf16split(x3[t] - m[t], h3, l3);
                store4bf(&Bh[c0 + t][r0], h0, h1, h2, h3);
                store4bf(&Bl[c0 + t][r0], l0, l1, l2, l3);
            }
        }
        __syncthreads();

        const int m = wid * 16 + (lane & 15);
        v16bf a_h = load_fragA(&Ah[m][0], lane);
        v16bf a_l = load_fragA(&Al[m][0], lane);
        v16bf b_h[4], b_l[4];
        #pragma unroll
        for (int t = 0; t < 4; ++t) {
            int n = t * 16 + (lane & 15);
            b_h[t] = load_fragB(&Bh[n][0], lane);
            b_l[t] = load_fragB(&Bl[n][0], lane);
        }
        #pragma unroll
        for (int t = 0; t < 4; ++t)
            acc[t] = __builtin_amdgcn_wmma_f32_16x16x32_bf16(false, a_h, false, b_h[t], (short)0, acc[t], false, false);
        #pragma unroll
        for (int t = 0; t < 4; ++t)
            acc[t] = __builtin_amdgcn_wmma_f32_16x16x32_bf16(false, a_h, false, b_l[t], (short)0, acc[t], false, false);
        #pragma unroll
        for (int t = 0; t < 4; ++t)
            acc[t] = __builtin_amdgcn_wmma_f32_16x16x32_bf16(false, a_l, false, b_h[t], (short)0, acc[t], false, false);
    }

    const float invN = 1.0f / (float)N;
    #pragma unroll
    for (int t = 0; t < 4; ++t) {
        #pragma unroll
        for (int r = 0; r < 8; ++r) {
            int i = i0 + wid * 16 + r + ((lane >> 4) << 3);
            int j = j0 + t * 16 + (lane & 15);
            float v = acc[t][r] * invN;
            if (i == j) v += BN_EPS;
            C[(size_t)i * F + j] = v;
        }
    }
}

// ----------------------------------------------- in-place Cholesky (lower)
__global__ __launch_bounds__(256) void cholesky_inplace(float* __restrict__ C, int F) {
    const int tid = threadIdx.x;
    for (int k = 0; k < F; ++k) {
        if (tid == 0) C[(size_t)k * F + k] = sqrtf(C[(size_t)k * F + k]);
        __syncthreads();
        const float inv = 1.0f / C[(size_t)k * F + k];
        for (int i = k + 1 + tid; i < F; i += 256)
            C[(size_t)i * F + k] *= inv;
        __syncthreads();
        for (int i = k + 1 + tid; i < F; i += 256) {
            const float lik = C[(size_t)i * F + k];
            for (int j = k + 1; j <= i; ++j)
                C[(size_t)i * F + j] -= lik * C[(size_t)j * F + k];
        }
        __syncthreads();
    }
}

// ------------------------------------------------- W = L^{-1} (lower tri)
__global__ __launch_bounds__(128) void tri_inverse(const float* __restrict__ L,
                                                   float* __restrict__ W, int F) {
    const int c0 = blockIdx.x * 128;
    const int j  = c0 + threadIdx.x;
    W[(size_t)j * F + j] = 1.0f / L[(size_t)j * F + j];
    for (int i = c0 + 1; i < F; ++i) {
        float s = 0.0f;
        for (int k = c0; k < i; ++k)                    // L[i][k] is wave-uniform
            s += L[(size_t)i * F + k] * W[(size_t)k * F + j];
        if (i > j)
            W[(size_t)i * F + j] = -s * (1.0f / L[(size_t)i * F + i]);
    }
}

// ------------------------------------------------- Y = xc * W^T + beta
// 128(n) x 64(j) tile per block. W^T is upper-triangular in k: k-chunks stop
// at j0+64. W rows are already [j][k] = the B[N][K] layout (no transpose);
// all staging stores are packed ds_store_b64.
__global__ __launch_bounds__(256) void whiten_gemm_wmma(const float* __restrict__ X,
                                                        const float* __restrict__ mu,
                                                        const float* __restrict__ W,
                                                        const float* __restrict__ beta,
                                                        float* __restrict__ Y,
                                                        int N, int F) {
    __shared__ __attribute__((aligned(16))) __bf16 Ah[128][LDS_LD], Al[128][LDS_LD];
    __shared__ __attribute__((aligned(16))) __bf16 Bh[64][LDS_LD],  Bl[64][LDS_LD];
    const int n0 = blockIdx.y * 128, j0 = blockIdx.x * 64;
    const int tid = threadIdx.x, lane = tid & 31, wid = tid >> 5;

    v8f acc[4] = {};
    const int kmax = (j0 + 64 < F) ? (j0 + 64) : F;

    for (int k0 = 0; k0 < kmax; k0 += 32) {
        __syncthreads();
        // stage A: xc tile [128 n][32 k], k-contiguous packed stores
        for (int idx = tid; idx < 1024; idx += 256) {
            int r = idx >> 3;               // n row 0..127
            int c = (idx & 7) << 2;         // k col 0..28
            f32x4 x = *(const f32x4*)&X[(size_t)(n0 + r) * F + k0 + c];
            f32x4 m = *(const f32x4*)&mu[k0 + c];
            __bf16 h0, h1, h2, h3, l0, l1, l2, l3;
            bf16split(x[0] - m[0], h0, l0);
            bf16split(x[1] - m[1], h1, l1);
            bf16split(x[2] - m[2], h2, l2);
            bf16split(x[3] - m[3], h3, l3);
            store4bf(&Ah[r][c], h0, h1, h2, h3);
            store4bf(&Al[r][c], l0, l1, l2, l3);
        }
        // stage B: W tile [64 j][32 k]
        for (int idx = tid; idx < 512; idx += 256) {
            int r = idx >> 3;               // j row 0..63
            int c = (idx & 7) << 2;
            f32x4 x = *(const f32x4*)&W[(size_t)(j0 + r) * F + k0 + c];
            __bf16 h0, h1, h2, h3, l0, l1, l2, l3;
            bf16split(x[0], h0, l0);
            bf16split(x[1], h1, l1);
            bf16split(x[2], h2, l2);
            bf16split(x[3], h3, l3);
            store4bf(&Bh[r][c], h0, h1, h2, h3);
            store4bf(&Bl[r][c], l0, l1, l2, l3);
        }
        __syncthreads();

        const int m = wid * 16 + (lane & 15);
        v16bf a_h = load_fragA(&Ah[m][0], lane);
        v16bf a_l = load_fragA(&Al[m][0], lane);
        v16bf b_h[4], b_l[4];
        #pragma unroll
        for (int t = 0; t < 4; ++t) {
            int n = t * 16 + (lane & 15);
            b_h[t] = load_fragB(&Bh[n][0], lane);
            b_l[t] = load_fragB(&Bl[n][0], lane);
        }
        #pragma unroll
        for (int t = 0; t < 4; ++t)
            acc[t] = __builtin_amdgcn_wmma_f32_16x16x32_bf16(false, a_h, false, b_h[t], (short)0, acc[t], false, false);
        #pragma unroll
        for (int t = 0; t < 4; ++t)
            acc[t] = __builtin_amdgcn_wmma_f32_16x16x32_bf16(false, a_h, false, b_l[t], (short)0, acc[t], false, false);
        #pragma unroll
        for (int t = 0; t < 4; ++t)
            acc[t] = __builtin_amdgcn_wmma_f32_16x16x32_bf16(false, a_l, false, b_h[t], (short)0, acc[t], false, false);
    }

    #pragma unroll
    for (int t = 0; t < 4; ++t) {
        #pragma unroll
        for (int r = 0; r < 8; ++r) {
            int n = n0 + wid * 16 + r + ((lane >> 4) << 3);
            int j = j0 + t * 16 + (lane & 15);
            Y[(size_t)n * F + j] = acc[t][r] + beta[j];
        }
    }
}

// ---------------------------------------------------------------------------
extern "C" void kernel_launch(void* const* d_in, const int* in_sizes, int n_in,
                              void* d_out, int out_size, void* d_ws, size_t ws_size,
                              hipStream_t stream) {
    const float* X    = (const float*)d_in[0];
    const float* beta = (const float*)d_in[3];
    float*       Y    = (float*)d_out;

    const int F = in_sizes[1];          // 1024
    const int N = in_sizes[0] / F;      // 32768
    const int G = 64;                   // mean reduction groups

    // workspace layout (floats): mu | partials | cov/L | W
    float* ws   = (float*)d_ws;
    float* mu   = ws;
    float* part = mu + F;
    float* cov  = part + (size_t)G * F;
    float* W    = cov + (size_t)F * F;

    // W upper triangle must be zero (also clears 0xAA poison)
    zero_f32<<<4096, 256, 0, stream>>>(W, (long)F * F);

    // 1) mean (two-stage, deterministic)
    mean_partial<<<dim3(F / 256, G), 256, 0, stream>>>(X, part, F, N / G);
    mean_reduce<<<(F + 255) / 256, 256, 0, stream>>>(part, mu, F, G, 1.0f / (float)N);

    // 2) cov = xc^T xc / N + eps I   (bf16x3 WMMA, lower-triangle blocks only)
    cov_syrk_wmma<<<dim3(F / 64, F / 128), 256, 0, stream>>>(X, mu, cov, N, F);

    // 3) L = chol(cov) in-place
    cholesky_inplace<<<1, 256, 0, stream>>>(cov, F);

    // 4) W = L^{-1}
    tri_inverse<<<F / 128, 128, 0, stream>>>(cov, W, F);

    // 5) Y = xc * W^T + beta        (bf16x3 WMMA, triangular k-bound)
    whiten_gemm_wmma<<<dim3(F / 64, N / 128), 256, 0, stream>>>(X, mu, W, beta, Y, N, F);
}
